// TPF_Encoder_34857954574856
// MI455X (gfx1250) — matrix operand order repaired and checked by reading, not compiled, not run on hardware
//
#include <hip/hip_runtime.h>

typedef __attribute__((ext_vector_type(16))) _Float16 v16h;
typedef __attribute__((ext_vector_type(8)))  _Float16 v8h;
typedef __attribute__((ext_vector_type(4)))  _Float16 v4h;
typedef __attribute__((ext_vector_type(8)))  float    v8f;

#define EPSLN 1e-5f

// ---------------------------------------------------------------------------
// Fragment loaders (layouts per CDNA5 ISA 7.12.2, wave32) — vectorized.
// ---------------------------------------------------------------------------

// A-matrix 16x32 f16 fragment from LDS (row-major, strideH halves per row).
// Per lane the v16h is two contiguous 8-half runs:
//   elements 0..7  -> K = kBase + hi*8      .. +7
//   elements 8..15 -> K = kBase + 16 + hi*8 .. +7
// => two 16B-aligned ds_load_b128.
__device__ inline v16h load_afrag(const _Float16* lds, int mBase, int kBase, int strideH) {
    int lane = threadIdx.x & 31;
    int hi   = lane >> 4;
    const _Float16* p = lds + (size_t)(mBase + (lane & 15)) * strideH + kBase + hi * 8;
    union { v16h v; v8h h[2]; } u;
    u.h[0] = *(const v8h*)p;
    u.h[1] = *(const v8h*)(p + 16);
    return u.v;
}

// B-matrix 32x16 f16 fragment from PRE-TRANSPOSED weights WT[N=128][K].
// lane: N = nBase + (lane&15); elements 0..7 -> K = kBase + hi*8 + e,
// elements 8..15 -> K = kBase + 16 + hi*8 + (e-8) => two global_load_b128.
__device__ inline v16h load_bfrag(const _Float16* __restrict__ WT, int K,
                                  int kBase, int nBase) {
    int lane = threadIdx.x & 31;
    int hi   = lane >> 4;
    const _Float16* p = WT + (size_t)(nBase + (lane & 15)) * K + kBase + hi * 8;
    union { v16h v; v8h h[2]; } u;
    u.h[0] = *(const v8h*)p;
    u.h[1] = *(const v8h*)(p + 16);
    return u.v;
}

// Store a wave's two 16x16 f32 accumulators to LDS out tile (64x128 f32).
// C layout: VGPR r holds M = r + hi*8, N = lane&15.
__device__ inline void store_acc(float* sC, int m_sub, int nBase, const v8f* acc) {
    int lane = threadIdx.x & 31;
    int hi   = lane >> 4;
    int n0   = lane & 15;
#pragma unroll
    for (int t = 0; t < 2; ++t)
#pragma unroll
        for (int r = 0; r < 8; ++r)
            sC[(m_sub * 16 + hi * 8 + r) * 128 + nBase + t * 16 + n0] = acc[t][r];
}

template <int NK>
__device__ inline void run_gemm(const v16h* aF, const _Float16* __restrict__ WT,
                                int K, int nBase, v8f* acc) {
#pragma unroll
    for (int kc = 0; kc < NK; ++kc) {
        v16h b0 = load_bfrag(WT, K, kc * 32, nBase);
        acc[0] = __builtin_amdgcn_wmma_f32_16x16x32_f16(false, aF[kc], false, b0,
                                                        (short)0, acc[0], false, false);
        v16h b1 = load_bfrag(WT, K, kc * 32, nBase + 16);
        acc[1] = __builtin_amdgcn_wmma_f32_16x16x32_f16(false, aF[kc], false, b1,
                                                        (short)0, acc[1], false, false);
    }
}

// ---------------------------------------------------------------------------
// Staging: gather 64 rows x 128 cols f32 -> f16 LDS tile (optionally indexed)
// one global_load_b128 + one ds_store_b64 per float4 slot.
// ---------------------------------------------------------------------------
__device__ inline void stage_gather(const float* __restrict__ tab,
                                    const int* __restrict__ idx,
                                    int base, int limit,
                                    _Float16* dst, int strideH) {
    int t = threadIdx.x;
#pragma unroll
    for (int j = 0; j < 4; ++j) {
        int v   = t + 512 * j;      // 0..2047 float4 slots
        int row = v >> 5;           // 32 float4 per row
        int c4  = v & 31;
        int e   = base + row;
        if (e >= limit) e = limit - 1;
        int r   = idx ? idx[e] : e;
        const float4 f = ((const float4*)(tab + (size_t)r * 128))[c4];
        v4h h = {(_Float16)f.x, (_Float16)f.y, (_Float16)f.z, (_Float16)f.w};
        *(v4h*)(dst + (size_t)row * strideH + c4 * 4) = h;
    }
}

// ---------------------------------------------------------------------------
// LayerNorm(+bias)+ReLU over 128 cols. Each wave owns rows 4w..4w+3; each
// lane owns 4 contiguous cols (lane*4..lane*4+3) => vector loads/stores.
// Reads f32 tile into registers, barrier, writes f16 (region may overlap).
// ---------------------------------------------------------------------------
__device__ inline void ln_relu_f16(const float* sC,
                                   const float* __restrict__ bias,
                                   const float* __restrict__ g,
                                   const float* __restrict__ be,
                                   _Float16* out /*64x128*/) {
    int wave = threadIdx.x >> 5, lane = threadIdx.x & 31;
    float4 bv = *(const float4*)(bias + lane * 4);
    float4 gv = *(const float4*)(g    + lane * 4);
    float4 ev = *(const float4*)(be   + lane * 4);
    float4 x[4];
#pragma unroll
    for (int r = 0; r < 4; ++r) {
        int row = wave * 4 + r;
        float4 c = *(const float4*)(sC + row * 128 + lane * 4);
        x[r] = make_float4(c.x + bv.x, c.y + bv.y, c.z + bv.z, c.w + bv.w);
    }
    __syncthreads();
#pragma unroll
    for (int r = 0; r < 4; ++r) {
        int row = wave * 4 + r;
        float s  = x[r].x + x[r].y + x[r].z + x[r].w;
        float s2 = x[r].x * x[r].x + x[r].y * x[r].y
                 + x[r].z * x[r].z + x[r].w * x[r].w;
#pragma unroll
        for (int off = 16; off > 0; off >>= 1) {
            s  += __shfl_xor(s,  off, 32);
            s2 += __shfl_xor(s2, off, 32);
        }
        float m  = s * (1.f / 128.f);
        float rs = rsqrtf(s2 * (1.f / 128.f) - m * m + EPSLN);
        v4h h;
        h[0] = (_Float16)fmaxf((x[r].x - m) * rs * gv.x + ev.x, 0.f);
        h[1] = (_Float16)fmaxf((x[r].y - m) * rs * gv.y + ev.y, 0.f);
        h[2] = (_Float16)fmaxf((x[r].z - m) * rs * gv.z + ev.z, 0.f);
        h[3] = (_Float16)fmaxf((x[r].w - m) * rs * gv.w + ev.w, 0.f);
        *(v4h*)(out + row * 128 + lane * 4) = h;
    }
}

// Final LN+ReLU -> scatter-add into agg[dst[row]] (edge kernel tail).
__device__ inline void ln_relu_scatter(const float* sC,
                                       const float* __restrict__ bias,
                                       const float* __restrict__ g,
                                       const float* __restrict__ be,
                                       const int* __restrict__ dst,
                                       int base, int E, float* agg) {
    int wave = threadIdx.x >> 5, lane = threadIdx.x & 31;
    float4 bv = *(const float4*)(bias + lane * 4);
    float4 gv = *(const float4*)(g    + lane * 4);
    float4 ev = *(const float4*)(be   + lane * 4);
#pragma unroll
    for (int r = 0; r < 4; ++r) {
        int row = wave * 4 + r;
        float4 c = *(const float4*)(sC + row * 128 + lane * 4);
        float4 x = make_float4(c.x + bv.x, c.y + bv.y, c.z + bv.z, c.w + bv.w);
        float s  = x.x + x.y + x.z + x.w;
        float s2 = x.x * x.x + x.y * x.y + x.z * x.z + x.w * x.w;
#pragma unroll
        for (int off = 16; off > 0; off >>= 1) {
            s  += __shfl_xor(s,  off, 32);
            s2 += __shfl_xor(s2, off, 32);
        }
        float m  = s * (1.f / 128.f);
        float rs = rsqrtf(s2 * (1.f / 128.f) - m * m + EPSLN);
        int rowg = base + row;
        if (rowg < E) {
            int d = dst[rowg];
            float* p = agg + (size_t)d * 128 + lane * 4;
            atomicAdd(p + 0, fmaxf((x.x - m) * rs * gv.x + ev.x, 0.f));
            atomicAdd(p + 1, fmaxf((x.y - m) * rs * gv.y + ev.y, 0.f));
            atomicAdd(p + 2, fmaxf((x.z - m) * rs * gv.z + ev.z, 0.f));
            atomicAdd(p + 3, fmaxf((x.w - m) * rs * gv.w + ev.w, 0.f));
        }
    }
}

// Final LN+ReLU -> dense f32 store (node kernel tail), one b128 per row/lane.
__device__ inline void ln_relu_store(const float* sC,
                                     const float* __restrict__ bias,
                                     const float* __restrict__ g,
                                     const float* __restrict__ be,
                                     int base, int N, float* out) {
    int wave = threadIdx.x >> 5, lane = threadIdx.x & 31;
    float4 bv = *(const float4*)(bias + lane * 4);
    float4 gv = *(const float4*)(g    + lane * 4);
    float4 ev = *(const float4*)(be   + lane * 4);
#pragma unroll
    for (int r = 0; r < 4; ++r) {
        int row = wave * 4 + r;
        float4 c = *(const float4*)(sC + row * 128 + lane * 4);
        float4 x = make_float4(c.x + bv.x, c.y + bv.y, c.z + bv.z, c.w + bv.w);
        float s  = x.x + x.y + x.z + x.w;
        float s2 = x.x * x.x + x.y * x.y + x.z * x.z + x.w * x.w;
#pragma unroll
        for (int off = 16; off > 0; off >>= 1) {
            s  += __shfl_xor(s,  off, 32);
            s2 += __shfl_xor(s2, off, 32);
        }
        float m  = s * (1.f / 128.f);
        float rs = rsqrtf(s2 * (1.f / 128.f) - m * m + EPSLN);
        int rowg = base + row;
        if (rowg < N) {
            float4 y;
            y.x = fmaxf((x.x - m) * rs * gv.x + ev.x, 0.f);
            y.y = fmaxf((x.y - m) * rs * gv.y + ev.y, 0.f);
            y.z = fmaxf((x.z - m) * rs * gv.z + ev.z, 0.f);
            y.w = fmaxf((x.w - m) * rs * gv.w + ev.w, 0.f);
            *(float4*)(out + (size_t)rowg * 128 + lane * 4) = y;
        }
    }
}

// ---------------------------------------------------------------------------
// Edge MLP: bh = relu(LN(relu(LN([h[src],bond] W1 + b1)) W2 + b2));
//           agg[dst] += bh          (fully fused, 64 edges per block)
// ---------------------------------------------------------------------------
__global__ __launch_bounds__(512, 1)
void edge_mlp(const float* __restrict__ hTab, const float* __restrict__ bond,
              const int* __restrict__ src, const int* __restrict__ dst, int E,
              const _Float16* __restrict__ W1, const float* __restrict__ b1,
              const float* __restrict__ g1, const float* __restrict__ be1,
              const _Float16* __restrict__ W2, const float* __restrict__ b2,
              const float* __restrict__ g2, const float* __restrict__ be2,
              float* agg) {
    __shared__ __align__(16) char smem[49152];
    _Float16* sA = (_Float16*)smem;              // [0,32K)  64x256 f16 input
    float*    sC = (float*)(smem + 16384);       // [16K,48K) 64x128 f32 out tile
    _Float16* sB = (_Float16*)(smem + 32768);    // [32K,48K) 64x128 f16 hidden

    int base = blockIdx.x * 64;
    stage_gather(hTab, src,     base, E, sA,       256);
    stage_gather(bond, nullptr, base, E, sA + 128, 256);
    __syncthreads();

    int wave  = threadIdx.x >> 5;
    int m_sub = wave & 3;
    int nBase = (wave >> 2) * 32;

    // GEMM1: K=256. Preload all A-frags so sC may overlay sA.
    v16h aF[8];
#pragma unroll
    for (int kc = 0; kc < 8; ++kc) aF[kc] = load_afrag(sA, m_sub * 16, kc * 32, 256);
    __syncthreads();
    v8f acc[2] = {};
    run_gemm<8>(aF, W1, 256, nBase, acc);
    store_acc(sC, m_sub, nBase, acc);
    __syncthreads();

    ln_relu_f16(sC, b1, g1, be1, sB);
    __syncthreads();

    // GEMM2: K=128
    v16h aG[4];
#pragma unroll
    for (int kc = 0; kc < 4; ++kc) aG[kc] = load_afrag(sB, m_sub * 16, kc * 32, 128);
    __syncthreads();
    v8f acc2[2] = {};
    run_gemm<4>(aG, W2, 128, nBase, acc2);
    store_acc(sC, m_sub, nBase, acc2);
    __syncthreads();

    ln_relu_scatter(sC, b2, g2, be2, dst, base, E, agg);
}

// ---------------------------------------------------------------------------
// Node MLP: f  = relu(LN([ax,agg] W1 + b1)); f = relu(LN(f W2 + b2));
//           out= relu(LN([ax,f] W3 + b3))   (64 nodes per block)
// ---------------------------------------------------------------------------
__global__ __launch_bounds__(512, 1)
void node_mlp(const float* __restrict__ ax, const float* __restrict__ agg, int N,
              const _Float16* __restrict__ W1, const float* __restrict__ b1,
              const float* __restrict__ g1, const float* __restrict__ be1,
              const _Float16* __restrict__ W2, const float* __restrict__ b2,
              const float* __restrict__ g2, const float* __restrict__ be2,
              const _Float16* __restrict__ W3, const float* __restrict__ b3,
              const float* __restrict__ g3, const float* __restrict__ be3,
              float* out) {
    __shared__ __align__(16) char smem[49152];
    _Float16* sAx = (_Float16*)smem;             // [0,16K)  persistent ax f16
    _Float16* sS  = (_Float16*)(smem + 16384);   // [16K,32K) agg f16 (transient)
    float*    sC  = (float*)(smem + 16384);      // [16K,48K) f32 out tile
    _Float16* sF  = (_Float16*)(smem + 32768);   // [32K,48K) f16 hidden

    int base = blockIdx.x * 64;
    stage_gather(ax,  nullptr, base, N, sAx, 128);
    stage_gather(agg, nullptr, base, N, sS,  128);
    __syncthreads();

    int wave  = threadIdx.x >> 5;
    int m_sub = wave & 3;
    int nBase = (wave >> 2) * 32;

    // GEMM1: K=256 over [ax | agg]
    v16h aF[8];
#pragma unroll
    for (int kc = 0; kc < 4; ++kc) {
        aF[kc]     = load_afrag(sAx, m_sub * 16, kc * 32, 128);
        aF[4 + kc] = load_afrag(sS,  m_sub * 16, kc * 32, 128);
    }
    __syncthreads();
    v8f acc[2] = {};
    run_gemm<8>(aF, W1, 256, nBase, acc);
    store_acc(sC, m_sub, nBase, acc);
    __syncthreads();
    ln_relu_f16(sC, b1, g1, be1, sF);
    __syncthreads();

    // GEMM2: K=128
    v16h aG[4];
#pragma unroll
    for (int kc = 0; kc < 4; ++kc) aG[kc] = load_afrag(sF, m_sub * 16, kc * 32, 128);
    __syncthreads();
    v8f acc2[2] = {};
    run_gemm<4>(aG, W2, 128, nBase, acc2);
    store_acc(sC, m_sub, nBase, acc2);
    __syncthreads();
    ln_relu_f16(sC, b2, g2, be2, sF);
    __syncthreads();

    // GEMM3: K=256 over [ax | f]
    v16h aH[8];
#pragma unroll
    for (int kc = 0; kc < 4; ++kc) {
        aH[kc]     = load_afrag(sAx, m_sub * 16, kc * 32, 128);
        aH[4 + kc] = load_afrag(sF,  m_sub * 16, kc * 32, 128);
    }
    __syncthreads();
    v8f acc3[2] = {};
    run_gemm<8>(aH, W3, 256, nBase, acc3);
    store_acc(sC, m_sub, nBase, acc3);
    __syncthreads();

    ln_relu_store(sC, b3, g3, be3, base, N, out);
}

// Weight f32 [S][K][128] -> f16 TRANSPOSED [S][128][K]
__global__ void transpose_to_f16(const float* __restrict__ in, _Float16* out, int K) {
    int i = blockIdx.x * 256 + threadIdx.x;        // element within slice
    int s = blockIdx.y;
    if (i < K * 128) {
        int k = i >> 7, n = i & 127;
        out[(size_t)s * 128 * K + (size_t)n * K + k] =
            (_Float16)in[(size_t)s * K * 128 + i];
    }
}

// ---------------------------------------------------------------------------
extern "C" void kernel_launch(void* const* d_in, const int* in_sizes, int n_in,
                              void* d_out, int out_size, void* d_ws, size_t ws_size,
                              hipStream_t stream) {
    (void)in_sizes; (void)n_in; (void)out_size; (void)ws_size;
    const float* h_top   = (const float*)d_in[0];   // [320000,128]
    const float* atom1   = (const float*)d_in[1];   // [80000,128]
    const float* atom0   = (const float*)d_in[2];   // [20000,128]
    const float* bond2   = (const float*)d_in[3];   // [320000,128]
    const float* bond1   = (const float*)d_in[4];   // [80000,128]
    const float* brW1    = (const float*)d_in[5];   // [2,256,128]
    const float* brB1    = (const float*)d_in[6];   // [2,128]
    const float* brG1    = (const float*)d_in[7];
    const float* brBe1   = (const float*)d_in[8];
    const float* brW2    = (const float*)d_in[9];   // [2,128,128]
    const float* brB2    = (const float*)d_in[10];
    const float* brG2    = (const float*)d_in[11];
    const float* brBe2   = (const float*)d_in[12];
    const float* sbW1    = (const float*)d_in[13];  // [2,256,128]
    const float* sbB1    = (const float*)d_in[14];
    const float* sbG1    = (const float*)d_in[15];
    const float* sbBe1   = (const float*)d_in[16];
    const float* sbW2    = (const float*)d_in[17];  // [2,128,128]
    const float* sbB2    = (const float*)d_in[18];
    const float* sbG2    = (const float*)d_in[19];
    const float* sbBe2   = (const float*)d_in[20];
    const float* hdW     = (const float*)d_in[21];  // [2,256,128]
    const float* hdB     = (const float*)d_in[22];
    const float* hdG     = (const float*)d_in[23];
    const float* hdBe    = (const float*)d_in[24];
    const int*   src2    = (const int*)d_in[25];
    const int*   dst2    = (const int*)d_in[26];
    const int*   src1    = (const int*)d_in[27];
    const int*   dst1    = (const int*)d_in[28];

    const int n2 = 320000, n1 = 80000, n0 = 20000;

    // Workspace layout (weights stored transposed: [S][128][K] f16)
    char* ws = (char*)d_ws;
    _Float16* fBrW1 = (_Float16*)(ws + 0);            // 2*128*256 halves
    _Float16* fBrW2 = (_Float16*)(ws + 131072);       // 2*128*128
    _Float16* fSbW1 = (_Float16*)(ws + 196608);       // 2*128*256
    _Float16* fSbW2 = (_Float16*)(ws + 327680);       // 2*128*128
    _Float16* fHdW  = (_Float16*)(ws + 393216);       // 2*128*256
    float* agg1 = (float*)(ws + 524288);                         // 80000*128 f32
    float* h1   = (float*)(ws + 524288 + (size_t)n1 * 128 * 4);  // 80000*128 f32
    float* agg0 = (float*)(ws + 524288 + 2 * (size_t)n1 * 128 * 4); // 20000*128

    // Convert + transpose weights to f16 (small; once per call, deterministic)
    transpose_to_f16<<<dim3(128, 2), 256, 0, stream>>>(brW1, fBrW1, 256);
    transpose_to_f16<<<dim3(64, 2),  256, 0, stream>>>(brW2, fBrW2, 128);
    transpose_to_f16<<<dim3(128, 2), 256, 0, stream>>>(sbW1, fSbW1, 256);
    transpose_to_f16<<<dim3(64, 2),  256, 0, stream>>>(sbW2, fSbW2, 128);
    transpose_to_f16<<<dim3(128, 2), 256, 0, stream>>>(hdW,  fHdW,  256);

    // ---- Level 0 (k=2): edges n2 -> nodes n1 ----
    hipMemsetAsync(agg1, 0, (size_t)n1 * 128 * 4, stream);
    edge_mlp<<<n2 / 64, 512, 0, stream>>>(
        h_top, bond2, src2, dst2, n2,
        fBrW1, brB1, brG1, brBe1,
        fBrW2, brB2, brG2, brBe2, agg1);
    node_mlp<<<n1 / 64, 512, 0, stream>>>(
        atom1, agg1, n1,
        fSbW1, sbB1, sbG1, sbBe1,
        fSbW2, sbB2, sbG2, sbBe2,
        fHdW,  hdB,  hdG,  hdBe, h1);

    // ---- Level 1 (k=1): edges n1 -> nodes n0, slice i=1 of weights ----
    hipMemsetAsync(agg0, 0, (size_t)n0 * 128 * 4, stream);
    edge_mlp<<<n1 / 64, 512, 0, stream>>>(
        h1, bond1, src1, dst1, n1,
        fBrW1 + 32768, brB1 + 128, brG1 + 128, brBe1 + 128,
        fBrW2 + 16384, brB2 + 128, brG2 + 128, brBe2 + 128, agg0);
    node_mlp<<<(n0 + 63) / 64, 512, 0, stream>>>(
        atom0, agg0, n0,
        fSbW1 + 32768, sbB1 + 128, sbG1 + 128, sbBe1 + 128,
        fSbW2 + 16384, sbB2 + 128, sbG2 + 128, sbBe2 + 128,
        fHdW  + 32768, hdB  + 128, hdG  + 128, hdBe + 128,
        (float*)d_out);
}
